// GCN_63393717289267
// MI455X (gfx1250) — matrix operand order repaired and checked by reading, not compiled, not run on hardware
//
#include <hip/hip_runtime.h>
#include <hip/hip_bf16.h>
#include <math.h>

// ---------------------------------------------------------------------------
// MI455X (gfx1250) implementation of the GCN+LSTM+GRU pipeline.
//   - Dense layers: wave32 WMMA (v_wmma_f32_16x16x32_f16), f16 in / f32 acc,
//     register-double-buffered K loop (loads overlap WMMA issue).
//   - Edge phase (L2-bandwidth-bound): fused gather + global_atomic_add_f32
//     segment sum; streamed operands use non-temporal loads so the L2-resident
//     h_msg gather table stays hot in the near caches.
// ---------------------------------------------------------------------------

typedef _Float16 v16h __attribute__((ext_vector_type(16)));
typedef _Float16 v8h  __attribute__((ext_vector_type(8)));
typedef float    v8f  __attribute__((ext_vector_type(8)));

#define N_DEG 6

// --------------------------- f32 -> f16 cast -------------------------------
__global__ __launch_bounds__(256)
void cast_f32_f16(const float* __restrict__ in, _Float16* __restrict__ out, int n) {
    int i = (blockIdx.x * 256 + threadIdx.x) * 4;
    if (i + 3 < n) {
        float4 v = *(const float4*)(in + i);
        out[i + 0] = (_Float16)v.x;
        out[i + 1] = (_Float16)v.y;
        out[i + 2] = (_Float16)v.z;
        out[i + 3] = (_Float16)v.w;
    } else {
        for (; i < n; ++i) out[i] = (_Float16)in[i];
    }
}

// --------------------------- WMMA GEMM (NT) --------------------------------
// C[M,Nout] = A[M,K] @ W[Nout,K]^T (+ bias[Nout]) (+= existing C if accumulate)
// block = 128 threads = 4 waves; wave computes a 16x64 strip (4 N-tiles,
// A fragment reused across the 4 WMMAs). grid = (Nout/256, M/16).
// K is a compile-time constant (128 or 256) so the loop fully unrolls and the
// next K-step's fragments are prefetched while the current WMMAs execute.
// Requires: M%16==0, Nout%256==0 (satisfied by this model).
template <int K>
__global__ __launch_bounds__(128)
void wmma_gemm_nt(const _Float16* __restrict__ A, const _Float16* __restrict__ W,
                  const float* __restrict__ bias, float* __restrict__ C,
                  int Nout, int accumulate)
{
    const int lane    = threadIdx.x & 31;
    const int wave    = threadIdx.x >> 5;
    const int half    = lane >> 4;   // lane group 0: lanes 0-15, group 1: 16-31
    const int l       = lane & 15;
    const int rowBase = blockIdx.y * 16;
    const int colBase = blockIdx.x * 256 + wave * 64;

    v8f acc[4];
    if (accumulate) {
#pragma unroll
        for (int t = 0; t < 4; ++t) {
            int col = colBase + t * 16 + l;
#pragma unroll
            for (int r = 0; r < 8; ++r)
                acc[t][r] = C[(size_t)(rowBase + half * 8 + r) * Nout + col];
        }
    } else {
#pragma unroll
        for (int t = 0; t < 4; ++t)
#pragma unroll
            for (int r = 0; r < 8; ++r) acc[t][r] = 0.0f;
    }

    // A fragment (16x32 f16, ISA 7.12.2): lane group `half` holds
    // K = {half*8 .. +7} and {16+half*8 .. +7} for row M = l.
    const _Float16* arow = A + (size_t)(rowBase + l) * K;
    // B fragment (32x16 f16): lane holds column n, K = {half*8 .. +7}
    // and {16+half*8 .. +7} (same order as A).
    const _Float16* wrow[4];
#pragma unroll
    for (int t = 0; t < 4; ++t)
        wrow[t] = W + (size_t)(colBase + t * 16 + l) * K;

    auto loadA = [&](int k) {
        v8h lo = *(const v8h*)(arow + k + half * 8);
        v8h hi = *(const v8h*)(arow + k + 16 + half * 8);
        v16h a;
#pragma unroll
        for (int e = 0; e < 8; ++e) { a[e] = lo[e]; a[8 + e] = hi[e]; }
        return a;
    };
    auto loadB = [&](int k, int t) {
        v8h lo = *(const v8h*)(wrow[t] + k + half * 8);
        v8h hi = *(const v8h*)(wrow[t] + k + 16 + half * 8);
        v16h b;
#pragma unroll
        for (int e = 0; e < 8; ++e) { b[e] = lo[e]; b[8 + e] = hi[e]; }
        return b;
    };

    // Software pipeline: fragments for step k are in registers while the
    // loads for step k+32 are in flight behind the 4 WMMAs.
    v16h a  = loadA(0);
    v16h b0 = loadB(0, 0), b1 = loadB(0, 1), b2 = loadB(0, 2), b3 = loadB(0, 3);

#pragma unroll
    for (int k = 0; k < K; k += 32) {
        v16h an = {}, bn0 = {}, bn1 = {}, bn2 = {}, bn3 = {};
        if (k + 32 < K) {
            an  = loadA(k + 32);
            bn0 = loadB(k + 32, 0);
            bn1 = loadB(k + 32, 1);
            bn2 = loadB(k + 32, 2);
            bn3 = loadB(k + 32, 3);
        }
        acc[0] = __builtin_amdgcn_wmma_f32_16x16x32_f16(false, a, false, b0,
                                                        (short)0, acc[0], false, false);
        acc[1] = __builtin_amdgcn_wmma_f32_16x16x32_f16(false, a, false, b1,
                                                        (short)0, acc[1], false, false);
        acc[2] = __builtin_amdgcn_wmma_f32_16x16x32_f16(false, a, false, b2,
                                                        (short)0, acc[2], false, false);
        acc[3] = __builtin_amdgcn_wmma_f32_16x16x32_f16(false, a, false, b3,
                                                        (short)0, acc[3], false, false);
        a = an; b0 = bn0; b1 = bn1; b2 = bn2; b3 = bn3;
    }

    // C/D layout: VGPR r, lane group half -> row = r + half*8, col = lane%16.
#pragma unroll
    for (int t = 0; t < 4; ++t) {
        int col = colBase + t * 16 + l;
        float bv = bias ? bias[col] : 0.0f;
#pragma unroll
        for (int r = 0; r < 8; ++r) {
            int row = rowBase + half * 8 + r;
            C[(size_t)row * Nout + col] = acc[t][r] + bv;
        }
    }
}

// ------------------------ fused GCN edge kernel ----------------------------
// Per edge e, feature quad q (64 quads of 4 floats = 256 feats):
//   m = degrees[e] @ W_edge^T  (+ masked gathers from h_msg[src])  * norm[src]
//   atomicAdd into accum[dst].
// degrees/src/dst are streamed exactly once -> non-temporal loads, keeping
// WGP$/L0 capacity for the heavily re-used h_msg gather table (L2-resident).
__global__ __launch_bounds__(256)
void gcn_edge(const float* __restrict__ h_msg,    // [N, 6*256]
              const float* __restrict__ degrees,  // [E, 6]
              const float* __restrict__ W_edge,   // [256, 6]
              const float* __restrict__ norm,     // [N]
              const int* __restrict__ src, const int* __restrict__ dst,
              float* __restrict__ accum,          // [N, 256], pre-zeroed
              int E)
{
    int t = blockIdx.x * 256 + threadIdx.x;
    int e = t >> 6;
    int q = t & 63;
    if (e >= E) return;

    int s = __builtin_nontemporal_load(src + e);
    int d = __builtin_nontemporal_load(dst + e);

    float deg[N_DEG];
#pragma unroll
    for (int i = 0; i < N_DEG; ++i)
        deg[i] = __builtin_nontemporal_load(degrees + (size_t)e * N_DEG + i);

    float m[4];
#pragma unroll
    for (int j = 0; j < 4; ++j) {
        int f = q * 4 + j;
        float acc = 0.0f;
#pragma unroll
        for (int i = 0; i < N_DEG; ++i) acc += deg[i] * W_edge[f * N_DEG + i];
        m[j] = acc;
    }

    const float* hrow = h_msg + (size_t)s * (N_DEG * 256);
#pragma unroll
    for (int i = 0; i < N_DEG; ++i) {
        if (deg[i] > 0.0f) {
            float4 v = *(const float4*)(hrow + i * 256 + q * 4);
            m[0] += v.x; m[1] += v.y; m[2] += v.z; m[3] += v.w;
        }
    }

    float ns = norm[s];
    float* arow = accum + (size_t)d * 256 + q * 4;
    atomicAdd(arow + 0, m[0] * ns);
    atomicAdd(arow + 1, m[1] * ns);
    atomicAdd(arow + 2, m[2] * ns);
    atomicAdd(arow + 3, m[3] * ns);
}

// ------------------------- pointwise kernels -------------------------------
__device__ __forceinline__ float sigm(float x) { return 1.0f / (1.0f + expf(-x)); }

__global__ __launch_bounds__(256)
void gcn_finish(const float* __restrict__ accum, const float* __restrict__ h_self,
                const float* __restrict__ bias,  // [256]
                const float* __restrict__ norm,  // [N]
                float* __restrict__ h_out, int total)
{
    int idx = blockIdx.x * 256 + threadIdx.x;
    if (idx >= total) return;
    int v = idx >> 8;
    int f = idx & 255;
    float x = accum[idx] * norm[v] + h_self[idx] + bias[f];
    h_out[idx] = x > 0.0f ? x : 0.0f;
}

__global__ __launch_bounds__(256)
void lstm_pointwise(const float* __restrict__ gates,  // [N, 4*256] (i,f,g,o)
                    const float* __restrict__ c0,
                    float* __restrict__ h_out, float* __restrict__ c_out,
                    int total)
{
    int idx = blockIdx.x * 256 + threadIdx.x;
    if (idx >= total) return;
    int v = idx >> 8;
    int f = idx & 255;
    const float* g = gates + (size_t)v * 1024;
    float ig = sigm(g[f]);
    float fg = sigm(g[256 + f]);
    float gg = tanhf(g[512 + f]);
    float og = sigm(g[768 + f]);
    float c2 = fg * c0[idx] + ig * gg;
    float h2 = og * tanhf(c2);
    h_out[idx] = h2;
    c_out[idx] = c2;
}

__global__ __launch_bounds__(256)
void gru_pointwise(const float* __restrict__ gi,   // [N, 3*256] (r,z,n)
                   const float* __restrict__ gh,   // [N, 3*256]
                   const float* __restrict__ hprev,
                   float* __restrict__ out, int total)
{
    int idx = blockIdx.x * 256 + threadIdx.x;
    if (idx >= total) return;
    int v = idx >> 8;
    int f = idx & 255;
    const float* a = gi + (size_t)v * 768;
    const float* b = gh + (size_t)v * 768;
    float r = sigm(a[f] + b[f]);
    float z = sigm(a[256 + f] + b[256 + f]);
    float n = tanhf(a[512 + f] + r * b[512 + f]);
    out[idx] = (1.0f - z) * n + z * hprev[idx];
}

// ------------------------------- launcher ----------------------------------
extern "C" void kernel_launch(void* const* d_in, const int* in_sizes, int n_in,
                              void* d_out, int out_size, void* d_ws, size_t ws_size,
                              hipStream_t stream)
{
    const float* features   = (const float*)d_in[0];   // [N,128]
    const float* hidden_gru = (const float*)d_in[1];   // [N,256]
    const float* h0         = (const float*)d_in[2];   // [N,256]
    const float* c0         = (const float*)d_in[3];   // [N,256]
    const float* degrees    = (const float*)d_in[4];   // [E,6]
    const float* norm       = (const float*)d_in[5];   // [N,1]
    const int*   src        = (const int*)d_in[6];     // [E]
    const int*   dst        = (const int*)d_in[7];     // [E]
    // params: gcn[0..3]{W_self,b_self,W_node,b_node,W_edge,bias},
    //         lstm{W_ih,W_hh,b_ih,b_hh}, gru{W_ih,W_hh,b_ih,b_hh}
    const int GCN0 = 8, LSTM0 = 8 + 4 * 6, GRU0 = LSTM0 + 4;

    const int N = in_sizes[5];           // 10000
    const int E = in_sizes[6];           // 160000
    const int total = N * 256;

    // ---- workspace layout (~98.1 MB) ----
    char* p = (char*)d_ws;
    float*    h_msg  = (float*)p;    p += (size_t)N * 1536 * sizeof(float); // reused for gates
    float*    h_self = (float*)p;    p += (size_t)N * 256 * sizeof(float);
    float*    accum  = (float*)p;    p += (size_t)N * 256 * sizeof(float);
    float*    h_cur  = (float*)p;    p += (size_t)N * 256 * sizeof(float);
    _Float16* a16    = (_Float16*)p; p += (size_t)N * 256 * sizeof(_Float16);
    _Float16* w16    = (_Float16*)p; p += (size_t)1536 * 256 * sizeof(_Float16);

    float* out_gru  = (float*)d_out;
    float* out_h    = out_gru + (size_t)N * 256;   // h_lstm
    float* out_c    = out_h   + (size_t)N * 256;   // c_lstm

    auto castTo = [&](const float* f32, _Float16* f16, int n) {
        int blocks = (n / 4 + 255) / 256;
        cast_f32_f16<<<blocks, 256, 0, stream>>>(f32, f16, n);
    };
    // C[M,Nout] = A[M,K] @ W[Nout,K]^T + bias  (+= C if acc)
    auto gemm = [&](const float* A, const float* W, const float* bias,
                    float* C, int M, int Nout, int K, bool acc) {
        castTo(A, a16, M * K);
        castTo(W, w16, Nout * K);
        dim3 g(Nout / 256, M / 16);
        if (K == 128)
            wmma_gemm_nt<128><<<g, 128, 0, stream>>>(a16, w16, bias, C, Nout, acc ? 1 : 0);
        else
            wmma_gemm_nt<256><<<g, 128, 0, stream>>>(a16, w16, bias, C, Nout, acc ? 1 : 0);
    };

    auto gcn_layer = [&](const float* h_in, int in_f, int li, float* h_out) {
        const float* W_self = (const float*)d_in[GCN0 + li * 6 + 0];
        const float* b_self = (const float*)d_in[GCN0 + li * 6 + 1];
        const float* W_node = (const float*)d_in[GCN0 + li * 6 + 2];
        const float* b_node = (const float*)d_in[GCN0 + li * 6 + 3];
        const float* W_edge = (const float*)d_in[GCN0 + li * 6 + 4];
        const float* bias   = (const float*)d_in[GCN0 + li * 6 + 5];

        gemm(h_in, W_node, b_node, h_msg, N, 1536, in_f, false);   // [N,1536]
        gemm(h_in, W_self, b_self, h_self, N, 256, in_f, false);   // [N,256]
        hipMemsetAsync(accum, 0, (size_t)total * sizeof(float), stream);
        int eblocks = (E * 64 + 255) / 256;
        gcn_edge<<<eblocks, 256, 0, stream>>>(h_msg, degrees, W_edge, norm,
                                              src, dst, accum, E);
        gcn_finish<<<(total + 255) / 256, 256, 0, stream>>>(accum, h_self, bias,
                                                            norm, h_out, total);
    };

    // ---- pipeline ----
    gcn_layer(features, 128, 0, h_cur);
    gcn_layer(h_cur,    256, 1, h_cur);

    // LSTM cell: gates = x@W_ih^T + b_ih + h0@W_hh^T + b_hh
    float* gates = h_msg;                                   // reuse [N,1024]
    gemm(h_cur, (const float*)d_in[LSTM0 + 0], (const float*)d_in[LSTM0 + 2],
         gates, N, 1024, 256, false);
    gemm(h0,    (const float*)d_in[LSTM0 + 1], (const float*)d_in[LSTM0 + 3],
         gates, N, 1024, 256, true);
    lstm_pointwise<<<(total + 255) / 256, 256, 0, stream>>>(gates, c0, out_h, out_c, total);

    gcn_layer(out_h, 256, 2, h_cur);
    gcn_layer(h_cur, 256, 3, h_cur);

    // GRU cell
    float* gi = h_msg;                                      // reuse [N,768]
    float* gh = h_msg + (size_t)N * 768;                    // reuse [N,768]
    gemm(h_cur,      (const float*)d_in[GRU0 + 0], (const float*)d_in[GRU0 + 2],
         gi, N, 768, 256, false);
    gemm(hidden_gru, (const float*)d_in[GRU0 + 1], (const float*)d_in[GRU0 + 3],
         gh, N, 768, 256, false);
    gru_pointwise<<<(total + 255) / 256, 256, 0, stream>>>(gi, gh, hidden_gru,
                                                           out_gru, total);
}
